// MultiHeadAttention_77653008711976
// MI455X (gfx1250) — matrix operand order repaired and checked by reading, not compiled, not run on hardware
//
#include <hip/hip_runtime.h>
#include <hip/hip_bf16.h>

// ---------------- problem constants ----------------
#define Bc   2
#define Sc   2048
#define HIDc 2048
#define NHc  32
#define HDc  64
#define MTOT (Bc * Sc)            // 4096
static const size_t QSZ = (size_t)Bc * Sc * HIDc;   // 8,388,608 elems
static const size_t WSZ = (size_t)HIDc * HIDc;      // 4,194,304 elems

typedef __attribute__((ext_vector_type(16))) _Float16 v16h;
typedef __attribute__((ext_vector_type(8)))  _Float16 v8h;
typedef __attribute__((ext_vector_type(8)))  float    v8f;
typedef __attribute__((ext_vector_type(4)))  float    v4f;

// ---------------- WMMA fragment helpers (layouts per CDNA5 ISA 7.12.2) ----

// A matrix 16x32 f16: lane L holds row M=L&15; halves 0..7 = K chunk at
// (L&16 ? 8 : 0), halves 8..15 = same + 16.
__device__ __forceinline__ v16h load_a_h(const _Float16* rowPtr, int lane, int kb) {
    int off = kb + ((lane & 16) ? 8 : 0);
    v8h lo = *(const v8h*)(rowPtr + off);
    v8h hi = *(const v8h*)(rowPtr + off + 16);
    return __builtin_shufflevector(lo, hi, 0,1,2,3,4,5,6,7,8,9,10,11,12,13,14,15);
}

// A fragment built from fp32 data (convert on load).
__device__ __forceinline__ v16h load_a_f(const float* rowPtr, int lane, int kb) {
    int off = kb + ((lane & 16) ? 8 : 0);
    const v4f* p0 = (const v4f*)(rowPtr + off);
    const v4f* p1 = (const v4f*)(rowPtr + off + 16);
    v4f f0 = p0[0], f1 = p0[1], f2 = p1[0], f3 = p1[1];
    v16h r;
#pragma unroll
    for (int i = 0; i < 4; ++i) {
        r[i]      = (_Float16)f0[i];
        r[4 + i]  = (_Float16)f1[i];
        r[8 + i]  = (_Float16)f2[i];
        r[12 + i] = (_Float16)f3[i];
    }
    return r;
}

// B matrix 32x16 f16: lane L holds column N=L&15; halves 0..7 = K chunk at
// (L&16 ? 8 : 0), halves 8..15 = same + 16.
__device__ __forceinline__ v16h load_b_h(const _Float16* rowPtr, int lane, int kb) {
    int off = kb + ((lane & 16) ? 8 : 0);
    v8h lo = *(const v8h*)(rowPtr + off);
    v8h hi = *(const v8h*)(rowPtr + off + 16);
    return __builtin_shufflevector(lo, hi, 0,1,2,3,4,5,6,7,8,9,10,11,12,13,14,15);
}

__device__ __forceinline__ v8f wmma16(v16h a, v16h b, v8f c) {
    return __builtin_amdgcn_wmma_f32_16x16x32_f16(false, a, false, b,
                                                  (short)0, c, false, false);
}

// ---------------- 0: fp32 -> f16 bulk convert ------------------------------
// 8 elements per thread, b128 in / b128 out.
__global__ void cvt_f32_to_f16_kernel(const float* __restrict__ src,
                                      _Float16* __restrict__ dst,
                                      unsigned n8) {
    unsigned i = blockIdx.x * 256u + threadIdx.x;
    if (i >= n8) return;
    size_t base = (size_t)i * 8;
    const v4f* p = (const v4f*)(src + base);
    v4f f0 = p[0], f1 = p[1];
    v8h r;
#pragma unroll
    for (int k = 0; k < 4; ++k) {
        r[k]     = (_Float16)f0[k];
        r[4 + k] = (_Float16)f1[k];
    }
    *(v8h*)(dst + base) = r;
}

// ---------------- 1/6: GEMM  Out[m,n] = sum_k A[m,k] * W[n,k] + bias[n] ----
// A: f16 row-major [M,2048]; W: f16 row-major [2048,2048] (pre-converted).
// Block tile 128x64, 256 thr = 8 waves laid out 4(m) x 2(n); each wave owns a
// 32x32 tile = 2x2 WMMA tiles -> 4 WMMA per k-step, k unrolled by 2.
__global__ void gemm_hh_kernel(const _Float16* __restrict__ A,
                               const _Float16* __restrict__ W,
                               const float* __restrict__ bias,
                               float* __restrict__ Out) {
    int tid = threadIdx.x, w = tid >> 5, lane = tid & 31;
    int m0 = blockIdx.x * 128 + (w >> 1) * 32;
    int n0 = blockIdx.y * 64 + (w & 1) * 32;
    const _Float16* ar0 = A + (size_t)(m0 + (lane & 15)) * HIDc;
    const _Float16* ar1 = ar0 + 16 * HIDc;
    const _Float16* wr0 = W + (size_t)(n0 + (lane & 15)) * HIDc;
    const _Float16* wr1 = wr0 + 16 * HIDc;
    v8f c00 = {}, c01 = {}, c10 = {}, c11 = {};
    for (int kb = 0; kb < HIDc; kb += 64) {
        // prefetch one 64-k super-step ahead into near caches (consumed by
        // this WGP next iteration -> WGP scope, max locality)
        __builtin_prefetch(ar0 + kb + 64, 0, 3);
        __builtin_prefetch(ar1 + kb + 64, 0, 3);
        __builtin_prefetch(wr0 + kb + 64, 0, 3);
        __builtin_prefetch(wr1 + kb + 64, 0, 3);
#pragma unroll
        for (int ks = 0; ks < 2; ++ks) {
            int k = kb + ks * 32;
            v16h a0 = load_a_h(ar0, lane, k);
            v16h a1 = load_a_h(ar1, lane, k);
            v16h b0 = load_b_h(wr0, lane, k);
            v16h b1 = load_b_h(wr1, lane, k);
            c00 = wmma16(a0, b0, c00);
            c01 = wmma16(a0, b1, c01);
            c10 = wmma16(a1, b0, c10);
            c11 = wmma16(a1, b1, c11);
        }
    }
    int ncol0 = n0 + (lane & 15);
    int ncol1 = ncol0 + 16;
    float bi0 = bias[ncol0], bi1 = bias[ncol1];
    int rb = (lane & 16) ? 8 : 0;
#pragma unroll
    for (int r = 0; r < 8; ++r) {
        size_t ro0 = (size_t)(m0 + rb + r) * HIDc;
        size_t ro1 = ro0 + (size_t)16 * HIDc;
        Out[ro0 + ncol0] = c00[r] + bi0;
        Out[ro0 + ncol1] = c01[r] + bi1;
        Out[ro1 + ncol0] = c10[r] + bi0;
        Out[ro1 + ncol1] = c11[r] + bi1;
    }
}

// ---------------- 2: RoPE + f16 convert -----------------------------------
// Thread per (b,h,s,i) i in [0,32): handles dim pair (i, i+32).
// qh/kh: [B][NH][S][HD] f16 ; vT: [B][NH][HD][S] f16 (transposed for P@V).
__global__ void rope_convert_kernel(const float* __restrict__ Qf,
                                    const float* __restrict__ Kf,
                                    const float* __restrict__ Vf,
                                    _Float16* __restrict__ qh,
                                    _Float16* __restrict__ kh,
                                    _Float16* __restrict__ vT) {
    unsigned idx = blockIdx.x * 256u + threadIdx.x;   // B*NH*S*32 = 4,194,304
    int i = idx & 31;
    int s = (idx >> 5) & (Sc - 1);
    int h = (idx >> 16) & (NHc - 1);
    int b = idx >> 21;
    // angle = s * 10000^(-i/32) = s * exp(-i * ln(1e4)/32)
    float ang = (float)s * __expf(-(float)i * (9.210340371976184f / 32.0f));
    float sn, cs;
    __sincosf(ang, &sn, &cs);
    size_t src = ((size_t)(b * Sc + s)) * HIDc + h * HDc + i;
    size_t dst = ((size_t)((b * NHc + h) * Sc + s)) * HDc + i;
    float q1 = Qf[src], q2 = Qf[src + 32];
    float k1 = Kf[src], k2 = Kf[src + 32];
    qh[dst]      = (_Float16)(q1 * cs - q2 * sn);
    qh[dst + 32] = (_Float16)(q2 * cs + q1 * sn);
    kh[dst]      = (_Float16)(k1 * cs - k2 * sn);
    kh[dst + 32] = (_Float16)(k2 * cs + k1 * sn);
    size_t vd = ((size_t)((b * NHc + h) * HDc + i)) * Sc + s;
    vT[vd]           = (_Float16)Vf[src];
    vT[vd + 32 * Sc] = (_Float16)Vf[src + 32];
}

// ---------------- 3: scores = (Q K^T)/sqrt(HD) + mask ---------------------
// Grid (S/64, S/64, B*NH). 64x64 tile/WG; K=HD=64 -> 2 WMMA steps.
__global__ void attn_scores_kernel(const _Float16* __restrict__ qh,
                                   const _Float16* __restrict__ kh,
                                   const float* __restrict__ mask,
                                   float* __restrict__ attn) {
    int tid = threadIdx.x, w = tid >> 5, lane = tid & 31;
    int bh = blockIdx.z;
    int b  = bh >> 5;
    int m0 = blockIdx.x * 64 + (w >> 1) * 16;
    int n0 = blockIdx.y * 64 + (w & 1) * 32;
    size_t base = (size_t)bh * Sc * HDc;
    const _Float16* qrow  = qh + base + (size_t)(m0 + (lane & 15)) * HDc;
    const _Float16* krow0 = kh + base + (size_t)(n0 + (lane & 15)) * HDc;
    const _Float16* krow1 = krow0 + 16 * HDc;
    v16h a0 = load_a_h(qrow, lane, 0);
    v16h a1 = load_a_h(qrow, lane, 32);
    v8f c0 = {}, c1 = {};
    c0 = wmma16(a0, load_b_h(krow0, lane, 0),  c0);
    c0 = wmma16(a1, load_b_h(krow0, lane, 32), c0);
    c1 = wmma16(a0, load_b_h(krow1, lane, 0),  c1);
    c1 = wmma16(a1, load_b_h(krow1, lane, 32), c1);
    int col0 = n0 + (lane & 15);
    float mk0 = mask[b * Sc + col0];
    float mk1 = mask[b * Sc + col0 + 16];
    int rb = (lane & 16) ? 8 : 0;
    float* arow = attn + (size_t)bh * Sc * Sc;
#pragma unroll
    for (int r = 0; r < 8; ++r) {
        size_t ro = (size_t)(m0 + rb + r) * Sc;
        arow[ro + col0]      = c0[r] * 0.125f + mk0;   // 1/sqrt(64)
        arow[ro + col0 + 16] = c1[r] * 0.125f + mk1;
    }
}

// ---------------- 4: row softmax (in place on attn) -----------------------
// One 256-thread WG per row of 2048; values held in registers, LDS tree
// reductions for max and sum.
__global__ void softmax_rows_kernel(float* __restrict__ attn) {
    __shared__ float red[256];
    float* p = attn + (size_t)blockIdx.x * Sc;
    int tid = threadIdx.x;
    float v[8];
#pragma unroll
    for (int i = 0; i < 8; ++i) v[i] = p[tid + 256 * i];
    float mx = v[0];
#pragma unroll
    for (int i = 1; i < 8; ++i) mx = fmaxf(mx, v[i]);
    red[tid] = mx;
    __syncthreads();
    for (int s = 128; s > 0; s >>= 1) {
        if (tid < s) red[tid] = fmaxf(red[tid], red[tid + s]);
        __syncthreads();
    }
    mx = red[0];
    __syncthreads();
    float sum = 0.f;
#pragma unroll
    for (int i = 0; i < 8; ++i) { v[i] = __expf(v[i] - mx); sum += v[i]; }
    red[tid] = sum;
    __syncthreads();
    for (int s = 128; s > 0; s >>= 1) {
        if (tid < s) red[tid] += red[tid + s];
        __syncthreads();
    }
    float inv = 1.f / red[0];
#pragma unroll
    for (int i = 0; i < 8; ++i) p[tid + 256 * i] = v[i] * inv;
}

// ---------------- 5: context = P @ V --------------------------------------
// Grid (S/64, B*NH). Wave w: m-tile w>>1, n-cols (w&1)*32 + {0,16}.
// A = attn fp32 (convert on load), B = vT f16 rows contiguous over K. K=2048.
__global__ void attn_context_kernel(const float* __restrict__ attn,
                                    const _Float16* __restrict__ vT,
                                    _Float16* __restrict__ ctx) {
    int tid = threadIdx.x, w = tid >> 5, lane = tid & 31;
    int bh = blockIdx.y;
    int b  = bh >> 5;
    int h  = bh & 31;
    int m1 = blockIdx.x * 64 + (w >> 1) * 16;
    int n0 = (w & 1) * 32;
    const float* prow = attn + ((size_t)bh * Sc + m1 + (lane & 15)) * Sc;
    const _Float16* v0 = vT + ((size_t)bh * HDc + n0 + (lane & 15)) * Sc;
    const _Float16* v1 = v0 + 16 * Sc;
    v8f c0 = {}, c1 = {};
    for (int kb = 0; kb < Sc; kb += 64) {
        __builtin_prefetch(prow + kb + 64, 0, 3);
        __builtin_prefetch(v0 + kb + 64, 0, 3);
        __builtin_prefetch(v1 + kb + 64, 0, 3);
#pragma unroll
        for (int ks = 0; ks < 2; ++ks) {
            int k = kb + ks * 32;
            v16h a  = load_a_f(prow, lane, k);
            v16h b0 = load_b_h(v0, lane, k);
            v16h b1 = load_b_h(v1, lane, k);
            c0 = wmma16(a, b0, c0);
            c1 = wmma16(a, b1, c1);
        }
    }
    int rb = (lane & 16) ? 8 : 0;
    int col = h * HDc + n0 + (lane & 15);
#pragma unroll
    for (int r = 0; r < 8; ++r) {
        size_t ro = ((size_t)b * Sc + m1 + rb + r) * HIDc;
        ctx[ro + col]      = (_Float16)c0[r];
        ctx[ro + col + 16] = (_Float16)c1[r];
    }
}

// ---------------- launch ---------------------------------------------------
extern "C" void kernel_launch(void* const* d_in, const int* in_sizes, int n_in,
                              void* d_out, int out_size, void* d_ws, size_t ws_size,
                              hipStream_t stream) {
    const float* hidden = (const float*)d_in[0];
    const float* mask   = (const float*)d_in[1];
    const float* Wq = (const float*)d_in[2];  const float* bq = (const float*)d_in[3];
    const float* Wk = (const float*)d_in[4];  const float* bk = (const float*)d_in[5];
    const float* Wv = (const float*)d_in[6];  const float* bv = (const float*)d_in[7];
    const float* Wo = (const float*)d_in[8];  const float* bo = (const float*)d_in[9];

    float* out  = (float*)d_out;
    float* attn = out + QSZ;                       // B*NH*S*S region (output #2)

    // Stage fp32 Q/K/V in the attn region (overwritten later by real scores).
    float* Qf = attn;
    float* Kf = attn + QSZ;
    float* Vf = attn + 2 * QSZ;

    // f16 tensors in workspace (~118 MB total).
    _Float16* qh   = (_Float16*)d_ws;
    _Float16* kh   = qh + QSZ;
    _Float16* vT   = kh + QSZ;
    _Float16* ctx  = vT + QSZ;
    _Float16* hidH = ctx + QSZ;
    _Float16* WqH  = hidH + QSZ;
    _Float16* WkH  = WqH + WSZ;
    _Float16* WvH  = WkH + WSZ;
    _Float16* WoH  = WvH + WSZ;

    dim3 blk(256);

    // 0) one-shot fp32 -> f16 conversion of activations and weights
    cvt_f32_to_f16_kernel<<<dim3((unsigned)(QSZ / 8 / 256)), blk, 0, stream>>>(hidden, hidH, (unsigned)(QSZ / 8));
    cvt_f32_to_f16_kernel<<<dim3((unsigned)(WSZ / 8 / 256)), blk, 0, stream>>>(Wq, WqH, (unsigned)(WSZ / 8));
    cvt_f32_to_f16_kernel<<<dim3((unsigned)(WSZ / 8 / 256)), blk, 0, stream>>>(Wk, WkH, (unsigned)(WSZ / 8));
    cvt_f32_to_f16_kernel<<<dim3((unsigned)(WSZ / 8 / 256)), blk, 0, stream>>>(Wv, WvH, (unsigned)(WSZ / 8));
    cvt_f32_to_f16_kernel<<<dim3((unsigned)(WSZ / 8 / 256)), blk, 0, stream>>>(Wo, WoH, (unsigned)(WSZ / 8));

    dim3 gProj(MTOT / 128, HIDc / 64);             // (32, 32)

    // 1) QKV projections (f16 x f16 WMMA, fp32 out + bias)
    gemm_hh_kernel<<<gProj, blk, 0, stream>>>(hidH, WqH, bq, Qf);
    gemm_hh_kernel<<<gProj, blk, 0, stream>>>(hidH, WkH, bk, Kf);
    gemm_hh_kernel<<<gProj, blk, 0, stream>>>(hidH, WvH, bv, Vf);

    // 2) RoPE + convert/transpose to f16
    rope_convert_kernel<<<dim3((Bc * NHc * Sc * 32) / 256), blk, 0, stream>>>(
        Qf, Kf, Vf, qh, kh, vT);

    // 3) scores (overwrites staging region with real scaled+masked scores)
    attn_scores_kernel<<<dim3(Sc / 64, Sc / 64, Bc * NHc), blk, 0, stream>>>(
        qh, kh, mask, attn);

    // 4) softmax over each of the B*NH*S rows
    softmax_rows_kernel<<<dim3(Bc * NHc * Sc), blk, 0, stream>>>(attn);

    // 5) context = P @ V
    attn_context_kernel<<<dim3(Sc / 64, Bc * NHc), blk, 0, stream>>>(attn, vT, ctx);

    // 6) output projection
    gemm_hh_kernel<<<gProj, blk, 0, stream>>>(ctx, WoH, bo, out);
}
